// DGCNN_32727650795899
// MI455X (gfx1250) — compile-verified, matrix operand order checked by reading, not run on hardware
//
#include <hip/hip_runtime.h>
#include <hip/hip_bf16.h>

typedef __attribute__((ext_vector_type(16))) _Float16     v16h;
typedef __attribute__((ext_vector_type(8)))  float        v8f;
typedef __attribute__((ext_vector_type(8)))  unsigned int v8u;

#define KNN 20
#define NPT 4096
#define NB  4
#define BN  (NB * NPT)         // 16384 points total
#define MBE (NPT * KNN)        // 81920 edges per batch

// ---------------------------------------------------------------------------
// Weight f32 -> f16 with zero padding of the channel dim up to Cpad
// ---------------------------------------------------------------------------
__global__ void cvtw_kernel(const float* __restrict__ src, _Float16* __restrict__ dst,
                            int O, int C, int Cpad) {
    int t = blockIdx.x * blockDim.x + threadIdx.x;
    if (t >= O * Cpad) return;
    int o = t / Cpad, c = t % Cpad;
    dst[t] = (c < C) ? (_Float16)src[o * C + c] : (_Float16)0.f;
}

// ---------------------------------------------------------------------------
// kNN: one thread per point; candidates tiled through LDS; register top-20.
// ---------------------------------------------------------------------------
template <int C, bool F32>
__global__ __launch_bounds__(128)
void knn_kernel(const void* __restrict__ srcv, int* __restrict__ idx) {
    const int b = blockIdx.y;
    const int n = blockIdx.x * 128 + threadIdx.x;
    __shared__ float tile[128 * C];

    const float*    sf = (const float*)srcv;
    const _Float16* sh = (const _Float16*)srcv;

    float own[C];
    float xx = 0.f;
#pragma unroll
    for (int c = 0; c < C; ++c) {
        float v = F32 ? sf[(size_t)b * C * NPT + (size_t)c * NPT + n]
                      : (float)sh[(size_t)c * BN + b * NPT + n];
        own[c] = v; xx += v * v;
    }

    float dk[KNN]; int ik[KNN];
#pragma unroll
    for (int j = 0; j < KNN; ++j) { dk[j] = -3.4e38f; ik[j] = 0; }

    for (int mt = 0; mt < NPT; mt += 128) {
        __syncthreads();
        for (int i = threadIdx.x; i < C * 128; i += 128) {
            int c = i >> 7, mm = i & 127;
            float v = F32 ? sf[(size_t)b * C * NPT + (size_t)c * NPT + mt + mm]
                          : (float)sh[(size_t)c * BN + b * NPT + mt + mm];
            tile[mm * C + c] = v;
        }
        __syncthreads();
        for (int mm = 0; mm < 128; ++mm) {
            float s = 0.f, x2 = 0.f;
#pragma unroll
            for (int c = 0; c < C; ++c) { float f = tile[mm * C + c]; s += own[c] * f; x2 += f * f; }
            float pd = 2.f * s - xx - x2;     // -||xi-xj||^2 ; self gives exactly 0
            if (pd > dk[KNN - 1]) {
                int pos = 0;
#pragma unroll
                for (int j = 0; j < KNN - 1; ++j) pos += (dk[j] >= pd) ? 1 : 0;
#pragma unroll
                for (int j = KNN - 1; j >= 0; --j) {
                    if (j > pos)        { dk[j] = dk[j - 1]; ik[j] = ik[j - 1]; }
                    else if (j == pos)  { dk[j] = pd; ik[j] = mt + mm; }
                }
            }
        }
    }
#pragma unroll
    for (int j = 0; j < KNN; ++j) idx[((size_t)(b * NPT + n)) * KNN + j] = ik[j];
}

// ---------------------------------------------------------------------------
// Edge feature build (per batch): F[2C, N*K] f16 = cat(nbr-ctr, ctr)
// ---------------------------------------------------------------------------
template <bool F32>
__global__ void graph_feat_kernel(const void* __restrict__ srcv, size_t cs,
                                  const int* __restrict__ idxb,
                                  _Float16* __restrict__ F, int C) {
    int t = blockIdx.x * blockDim.x + threadIdx.x;   // edge id in [0, N*K)
    if (t >= MBE) return;
    int n = t / KNN;
    int j = idxb[t];
    const float*    sf = (const float*)srcv;
    const _Float16* sh = (const _Float16*)srcv;
    for (int c = 0; c < C; ++c) {
        float ctr = F32 ? sf[c * cs + n] : (float)sh[c * cs + n];
        float nbr = F32 ? sf[c * cs + j] : (float)sh[c * cs + j];
        F[(size_t)c * MBE + t]       = (_Float16)(nbr - ctr);
        F[(size_t)(C + c) * MBE + t] = (_Float16)ctr;
    }
}

// ---------------------------------------------------------------------------
// Fused GEMM + BN + LeakyReLU via v_wmma_f32_16x16x32_f16.
// Y[O,M] = lrelu( (W[O,Cpad] . X[Creal,M]) * g[o]*rsqrt(1+eps) + b[o] )
// Block = 4 waves (128 thr); block tile 64(O) x 64(M); waves 2x2, each wave a
// 32x32 register tile (2 A-frags x 2 B-frags -> 4 WMMAs per 32-deep K chunk).
// O % 64 == 0, M % 64 == 0, Cpad % 32 == 0 (rows >= Creal read as zero).
// ---------------------------------------------------------------------------
__global__ __launch_bounds__(128)
void gemm_wmma_kernel(const _Float16* __restrict__ W, const _Float16* __restrict__ X,
                      _Float16* __restrict__ Y,
                      const float* __restrict__ gam, const float* __restrict__ bet,
                      int O, int Creal, int Cpad, int M) {
    __shared__ unsigned int ldsW[64 * 16];   // [o][c] row-major f16 (64 x 32), uint pairs
    __shared__ unsigned int ldsX[64 * 16];   // [m][c] transposed f16 (64 x 32), uint pairs

    const int t    = threadIdx.x;
    const int lane = t & 31;
    const int wav  = t >> 5;                 // 0..3
    const int hl   = lane >> 4;              // lane half (K-group select)
    const int row  = lane & 15;              // A row / B col within a 16x16 tile
    const int wo   = (wav & 1) << 5;         // wave O offset in block tile
    const int wm   = (wav >> 1) << 5;        // wave M offset in block tile
    const int m0 = blockIdx.x * 64;
    const int o0 = blockIdx.y * 64;

    v8f acc[2][2] = {{{}, {}}, {{}, {}}};

    for (int c0 = 0; c0 < Cpad; c0 += 32) {
        __syncthreads();
        // W tile: 64 rows x 32 f16 = 1024 uints; 128 threads x 2 uint4 loads.
#pragma unroll
        for (int p = 0; p < 2; ++p) {
            int uidx = p * 512 + t * 4;            // uint index, multiple of 4
            int o = uidx >> 4, cu = uidx & 15;     // row, uint-col
            uint4 v4 = *(const uint4*)(W + (size_t)(o0 + o) * Cpad + c0 + (cu << 1));
            *(uint4*)&ldsW[uidx] = v4;
        }
        // X tile: 32(c) x 64(m), stored transposed [m][c]; zero-fill c >= Creal.
#pragma unroll
        for (int p = 0; p < 4; ++p) {
            int li = p * 128 + t;                  // 512 chunks of 4 f16 along m
            int c = li >> 4, mq = (li & 15) << 2;
            unsigned int lo = 0u, hi = 0u;
            if (c0 + c < Creal) {
                uint2 v2 = *(const uint2*)(X + (size_t)(c0 + c) * M + m0 + mq);
                lo = v2.x; hi = v2.y;
            }
            unsigned short* lx = (unsigned short*)ldsX;
            lx[(mq + 0) * 32 + c] = (unsigned short)(lo & 0xffffu);
            lx[(mq + 1) * 32 + c] = (unsigned short)(lo >> 16);
            lx[(mq + 2) * 32 + c] = (unsigned short)(hi & 0xffffu);
            lx[(mq + 3) * 32 + c] = (unsigned short)(hi >> 16);
        }
        __syncthreads();

        // ISA 7.12.2 16-bit layout: VGPR vv holds K = (vv<4?0:16) + hl*8 + 2*(vv&3)
        v8u au0, au1, bu0, bu1;
#pragma unroll
        for (int vv = 0; vv < 8; ++vv) {
            int kb2 = ((vv & 4) << 1) + (hl << 2) + (vv & 3);   // uint index in row
            au0[vv] = ldsW[(wo + row) * 16 + kb2];
            au1[vv] = ldsW[(wo + 16 + row) * 16 + kb2];
            bu0[vv] = ldsX[(wm + row) * 16 + kb2];
            bu1[vv] = ldsX[(wm + 16 + row) * 16 + kb2];
        }
        v16h A0 = __builtin_bit_cast(v16h, au0);
        v16h A1 = __builtin_bit_cast(v16h, au1);
        v16h B0 = __builtin_bit_cast(v16h, bu0);
        v16h B1 = __builtin_bit_cast(v16h, bu1);
        acc[0][0] = __builtin_amdgcn_wmma_f32_16x16x32_f16(false, A0, false, B0, (short)0, acc[0][0], false, false);
        acc[0][1] = __builtin_amdgcn_wmma_f32_16x16x32_f16(false, A0, false, B1, (short)0, acc[0][1], false, false);
        acc[1][0] = __builtin_amdgcn_wmma_f32_16x16x32_f16(false, A1, false, B0, (short)0, acc[1][0], false, false);
        acc[1][1] = __builtin_amdgcn_wmma_f32_16x16x32_f16(false, A1, false, B1, (short)0, acc[1][1], false, false);
    }

    const float rs = rsqrtf(1.0f + 1e-5f);
#pragma unroll
    for (int i = 0; i < 2; ++i) {
#pragma unroll
        for (int j = 0; j < 2; ++j) {
#pragma unroll
            for (int r = 0; r < 8; ++r) {
                int og = o0 + wo + i * 16 + r + (hl << 3);
                int mg = m0 + wm + j * 16 + row;
                float y = acc[i][j][r] * (gam[og] * rs) + bet[og];
                y = (y >= 0.f) ? y : 0.2f * y;
                Y[(size_t)og * M + mg] = (_Float16)y;
            }
        }
    }
}

// ---------------------------------------------------------------------------
// Max over K neighbors (per batch) into xcat rows [row0, row0+64)
// ---------------------------------------------------------------------------
__global__ void maxk_kernel(const _Float16* __restrict__ Y, _Float16* __restrict__ xcat,
                            int row0, int b) {
    int t = blockIdx.x * blockDim.x + threadIdx.x;
    if (t >= 64 * NPT) return;
    int o = t / NPT, n = t % NPT;
    const _Float16* p = Y + (size_t)o * MBE + (size_t)n * KNN;
    float m = -3.4e38f;
    for (int k = 0; k < KNN; ++k) { float v = (float)p[k]; m = v > m ? v : m; }
    xcat[(size_t)(row0 + o) * BN + b * NPT + n] = (_Float16)m;
}

// Global max of h6 over N per (channel, batch)
__global__ void gmax_kernel(const _Float16* __restrict__ h6, _Float16* __restrict__ gmx) {
    int o = blockIdx.x, b = blockIdx.y;
    __shared__ float red[256];
    float m = -3.4e38f;
    for (int n = threadIdx.x; n < NPT; n += 256) {
        float v = (float)h6[(size_t)o * BN + b * NPT + n];
        m = v > m ? v : m;
    }
    red[threadIdx.x] = m; __syncthreads();
    for (int s = 128; s > 0; s >>= 1) {
        if (threadIdx.x < s) { float v = red[threadIdx.x + s]; if (v > red[threadIdx.x]) red[threadIdx.x] = v; }
        __syncthreads();
    }
    if (threadIdx.x == 0) gmx[o * NB + b] = (_Float16)red[0];
}

// Build hcat [1216, BN]: rows 0..1023 = gmx broadcast, rows 1024.. = xcat
__global__ void hcat_kernel(const _Float16* __restrict__ gmx, const _Float16* __restrict__ xcat,
                            _Float16* __restrict__ hcat) {
    size_t t = (size_t)blockIdx.x * blockDim.x + threadIdx.x;
    if (t >= (size_t)1216 * BN) return;
    int r = (int)(t / BN), bn = (int)(t % BN), b = bn / NPT;
    hcat[t] = (r < 1024) ? gmx[r * NB + b] : xcat[(size_t)(r - 1024) * BN + bn];
}

// Final 8x256 conv, f32 output in reference layout [B, 8, N]
__global__ void w9_kernel(const float* __restrict__ W9, const _Float16* __restrict__ h8,
                          float* __restrict__ out) {
    int t = blockIdx.x * blockDim.x + threadIdx.x;
    if (t >= NB * 8 * NPT) return;
    int n = t % NPT, o = (t / NPT) % 8, b = t / (8 * NPT);
    float acc = 0.f;
    for (int c = 0; c < 256; ++c)
        acc += W9[o * 256 + c] * (float)h8[(size_t)c * BN + b * NPT + n];
    out[t] = acc;
}

// ---------------------------------------------------------------------------
extern "C" void kernel_launch(void* const* d_in, const int* in_sizes, int n_in,
                              void* d_out, int out_size, void* d_ws, size_t ws_size,
                              hipStream_t stream) {
    (void)in_sizes; (void)n_in; (void)out_size; (void)ws_size;
    const float* x = (const float*)d_in[0];
    const float* Wf[10];
    for (int i = 1; i <= 9; ++i) Wf[i] = (const float*)d_in[i];
    const float *g[9], *bb[9];
    for (int i = 1; i <= 8; ++i) {
        g[i]  = (const float*)d_in[10 + (i - 1) * 2];
        bb[i] = (const float*)d_in[10 + (i - 1) * 2 + 1];
    }

    char* ws = (char*)d_ws;
    size_t off = 0;
    auto alloc = [&](size_t bytes) -> char* {
        off = (off + 255) & ~(size_t)255;
        char* p = ws + off; off += bytes; return p;
    };
    _Float16* w1p  = (_Float16*)alloc((size_t)64 * 32 * 2);
    _Float16* w2   = (_Float16*)alloc((size_t)64 * 64 * 2);
    _Float16* w3   = (_Float16*)alloc((size_t)64 * 128 * 2);
    _Float16* w4   = (_Float16*)alloc((size_t)64 * 64 * 2);
    _Float16* w5   = (_Float16*)alloc((size_t)64 * 128 * 2);
    _Float16* w6   = (_Float16*)alloc((size_t)1024 * 192 * 2);
    _Float16* w7   = (_Float16*)alloc((size_t)512 * 1216 * 2);
    _Float16* w8   = (_Float16*)alloc((size_t)256 * 512 * 2);
    int*      idx  = (int*)alloc((size_t)NB * NPT * KNN * 4);
    _Float16* F    = (_Float16*)alloc((size_t)128 * MBE * 2);   // per-batch edge features
    _Float16* Ya   = (_Float16*)alloc((size_t)64 * MBE * 2);    // per-batch conv temp
    _Float16* xcat = (_Float16*)alloc((size_t)192 * BN * 2);    // [x1;x2;x3]
    _Float16* gmx  = (_Float16*)alloc((size_t)1024 * NB * 2);
    _Float16* h6   = (_Float16*)alloc((size_t)1024 * BN * 2);
    _Float16* hcat = (_Float16*)alloc((size_t)1216 * BN * 2);
    _Float16* h7   = h6;                       // alias: h6 dead after gmax
    _Float16* h8   = h6 + (size_t)512 * BN;    // alias: disjoint from h7

    auto cvt = [&](const float* s, _Float16* d, int O, int C, int Cpad) {
        int n = O * Cpad;
        cvtw_kernel<<<(n + 255) / 256, 256, 0, stream>>>(s, d, O, C, Cpad);
    };
    cvt(Wf[1], w1p, 64, 18, 32);
    cvt(Wf[2], w2, 64, 64, 64);
    cvt(Wf[3], w3, 64, 128, 128);
    cvt(Wf[4], w4, 64, 64, 64);
    cvt(Wf[5], w5, 64, 128, 128);
    cvt(Wf[6], w6, 1024, 192, 192);
    cvt(Wf[7], w7, 512, 1216, 1216);
    cvt(Wf[8], w8, 256, 512, 512);

    auto gemm = [&](const _Float16* W, const _Float16* X, _Float16* Y,
                    const float* gam, const float* bet, int O, int Creal, int Cpad, int M) {
        dim3 grid(M / 64, O / 64);
        gemm_wmma_kernel<<<grid, 128, 0, stream>>>(W, X, Y, gam, bet, O, Creal, Cpad, M);
    };

    // ---- stage 1: C=9 input ----
    knn_kernel<9, true><<<dim3(NPT / 128, NB), 128, 0, stream>>>(x, idx);
    for (int b = 0; b < NB; ++b) {
        graph_feat_kernel<true><<<(MBE + 255) / 256, 256, 0, stream>>>(
            x + (size_t)b * 9 * NPT, (size_t)NPT, idx + (size_t)b * NPT * KNN, F, 9);
        gemm(w1p, F, Ya, g[1], bb[1], 64, 18, 32, MBE);
        gemm(w2, Ya, F, g[2], bb[2], 64, 64, 64, MBE);
        maxk_kernel<<<(64 * NPT + 255) / 256, 256, 0, stream>>>(F, xcat, 0, b);
    }
    // ---- stage 2: over x1 ----
    knn_kernel<64, false><<<dim3(NPT / 128, NB), 128, 0, stream>>>(xcat, idx);
    for (int b = 0; b < NB; ++b) {
        graph_feat_kernel<false><<<(MBE + 255) / 256, 256, 0, stream>>>(
            xcat + (size_t)b * NPT, (size_t)BN, idx + (size_t)b * NPT * KNN, F, 64);
        gemm(w3, F, Ya, g[3], bb[3], 64, 128, 128, MBE);
        gemm(w4, Ya, F, g[4], bb[4], 64, 64, 64, MBE);
        maxk_kernel<<<(64 * NPT + 255) / 256, 256, 0, stream>>>(F, xcat, 64, b);
    }
    // ---- stage 3: over x2 ----
    knn_kernel<64, false><<<dim3(NPT / 128, NB), 128, 0, stream>>>(xcat + (size_t)64 * BN, idx);
    for (int b = 0; b < NB; ++b) {
        graph_feat_kernel<false><<<(MBE + 255) / 256, 256, 0, stream>>>(
            xcat + (size_t)64 * BN + (size_t)b * NPT, (size_t)BN,
            idx + (size_t)b * NPT * KNN, F, 64);
        gemm(w5, F, Ya, g[5], bb[5], 64, 128, 128, MBE);
        maxk_kernel<<<(64 * NPT + 255) / 256, 256, 0, stream>>>(Ya, xcat, 128, b);
    }
    // ---- head ----
    gemm(w6, xcat, h6, g[6], bb[6], 1024, 192, 192, BN);
    gmax_kernel<<<dim3(1024, NB), 256, 0, stream>>>(h6, gmx);
    {
        size_t n = (size_t)1216 * BN;
        hcat_kernel<<<(unsigned)((n + 255) / 256), 256, 0, stream>>>(gmx, xcat, hcat);
    }
    gemm(w7, hcat, h7, g[7], bb[7], 512, 1216, 1216, BN);
    gemm(w8, h7, h8, g[8], bb[8], 256, 512, 512, BN);
    w9_kernel<<<(NB * 8 * NPT + 255) / 256, 256, 0, stream>>>(Wf[9], h8, (float*)d_out);
}
